// PointNet2Encoder_24850680775353
// MI455X (gfx1250) — compile-verified, matrix operand order checked by reading, not run on hardware
//
#include <hip/hip_runtime.h>
#include <cstdint>

// ---------------------------------------------------------------------------
// PointNet++ (3x SetAbstraction) encoder for gfx1250 (MI455X), wave32 + WMMA.
// ---------------------------------------------------------------------------

typedef __attribute__((ext_vector_type(16))) __bf16 bf16x16;
typedef __attribute__((ext_vector_type(8)))  __bf16 bf16x8;
typedef __attribute__((ext_vector_type(8)))  float  v8f;

static constexpr int BATCH = 8;
static constexpr int N0    = 4096;
static constexpr int M1s   = 2048, M2s = 512, M3s = 512;
static constexpr int KNBR  = 64;

// ---------------------------------------------------------------------------
// Pack f32 weight matrix [Kreal x Ncols] (row-major) into bf16 WMMA B-operand
// fragments for v_wmma_f32_16x16x32_bf16.  Fragment layout:
//   pk[((kt*NT + nt)*32 + lane)*16 + e]
// lane<16  holds column n = nt*16+lane,  K = {kt*32+e (e<8), kt*32+16+(e-8)}
// lane>=16 holds column n = nt*16+lane-16, same +8 K offset (mirrors A layout).
// K rows beyond Kreal are zero padding.
// ---------------------------------------------------------------------------
__global__ void pack_w_kernel(const float* __restrict__ w, __bf16* __restrict__ pk,
                              int Kreal, int KP, int Ncols) {
    int NT = Ncols >> 4;
    int total = (KP >> 5) * NT * 32 * 16;
    for (int idx = blockIdx.x * blockDim.x + threadIdx.x; idx < total;
         idx += gridDim.x * blockDim.x) {
        int e    = idx & 15;
        int lane = (idx >> 4) & 31;
        int f    = idx >> 9;              // fragment = kt*NT + nt
        int nt   = f % NT;
        int kt   = f / NT;
        int kk   = (e < 8 ? e : e + 8) + ((lane & 16) ? 8 : 0);
        int k    = kt * 32 + kk;
        int n    = nt * 16 + (lane & 15);
        float v  = (k < Kreal) ? w[k * Ncols + n] : 0.0f;
        pk[idx]  = (__bf16)v;
    }
}

// ---------------------------------------------------------------------------
// Furthest point sampling.  One block (256 threads) per batch element.
// Positions + running min-d2 are register-resident (N/256 per thread);
// argmax via wave32 shfl reduction + 8-wave LDS combine.  First-max tiebreak
// (lower index wins) to match jnp.argmax.
// ---------------------------------------------------------------------------
template <int N, int M>
__global__ void fps_kernel(const float* __restrict__ pos, float* __restrict__ pos_out) {
    constexpr int NT  = 256;
    constexpr int PPT = N / NT;
    const int b = blockIdx.x;
    const int t = threadIdx.x;
    const float* pb = pos + (size_t)b * N * 3;
    float* po = pos_out + (size_t)b * M * 3;

    __shared__ float s_red_v[8];
    __shared__ int   s_red_i[8];
    __shared__ float s_next[3];
    __shared__ int   s_nxt;

    float px[PPT], py[PPT], pz[PPT], md[PPT];
    const float x0 = pb[0], y0 = pb[1], z0 = pb[2];
#pragma unroll
    for (int j = 0; j < PPT; ++j) {
        int i = t + j * NT;
        float x = pb[i * 3 + 0], y = pb[i * 3 + 1], z = pb[i * 3 + 2];
        px[j] = x; py[j] = y; pz[j] = z;
        float dx = x - x0, dy = y - y0, dz = z - z0;
        md[j] = dx * dx + dy * dy + dz * dz;
    }
    if (t == 0) { po[0] = x0; po[1] = y0; po[2] = z0; }

    const int lane = t & 31, wv = t >> 5;
    for (int s = 1; s < M; ++s) {
        // per-thread argmax over its PPT points (strided ownership: i = t + j*NT)
        float bv = -1.0f; int bi = 0x7fffffff;
#pragma unroll
        for (int j = 0; j < PPT; ++j) {
            int i = t + j * NT;
            float v = md[j];
            if (v > bv || (v == bv && i < bi)) { bv = v; bi = i; }
        }
        // wave32 reduction
#pragma unroll
        for (int off = 16; off > 0; off >>= 1) {
            float ov = __shfl_xor(bv, off, 32);
            int   oi = __shfl_xor(bi, off, 32);
            if (ov > bv || (ov == bv && oi < bi)) { bv = ov; bi = oi; }
        }
        if (lane == 0) { s_red_v[wv] = bv; s_red_i[wv] = bi; }
        __syncthreads();
        if (t == 0) {
            float fv = s_red_v[0]; int fi = s_red_i[0];
#pragma unroll
            for (int w = 1; w < 8; ++w) {
                float ov = s_red_v[w]; int oi = s_red_i[w];
                if (ov > fv || (ov == fv && oi < fi)) { fv = ov; fi = oi; }
            }
            s_nxt = fi;
        }
        __syncthreads();
        const int nxt = s_nxt;
        if ((nxt & (NT - 1)) == t) {        // owner thread broadcasts + writes out
            int j = nxt / NT;
            s_next[0] = px[j]; s_next[1] = py[j]; s_next[2] = pz[j];
            po[s * 3 + 0] = px[j]; po[s * 3 + 1] = py[j]; po[s * 3 + 2] = pz[j];
        }
        __syncthreads();
        const float nx = s_next[0], ny = s_next[1], nz = s_next[2];
#pragma unroll
        for (int j = 0; j < PPT; ++j) {
            float dx = px[j] - nx, dy = py[j] - ny, dz = pz[j] - nz;
            md[j] = fminf(md[j], dx * dx + dy * dy + dz * dz);
        }
    }
}

// ---------------------------------------------------------------------------
// Ball query: one wave32 per query, ballot/popc stream compaction of the
// first <=64 in-radius source points.
// ---------------------------------------------------------------------------
__global__ void ballq_kernel(const float* __restrict__ pos_src,
                             const float* __restrict__ pos_q,
                             int* __restrict__ nbr, int* __restrict__ cnt,
                             int N, int M, float r2) {
    const int lane = threadIdx.x & 31;
    const int wv   = threadIdx.x >> 5;
    const int gq   = blockIdx.x * 8 + wv;           // over BATCH*M
    if (gq >= BATCH * M) return;
    const int b = gq / M, q = gq - b * M;
    const float* ps = pos_src + (size_t)b * N * 3;
    const float qx = pos_q[((size_t)b * M + q) * 3 + 0];
    const float qy = pos_q[((size_t)b * M + q) * 3 + 1];
    const float qz = pos_q[((size_t)b * M + q) * 3 + 2];
    int* nb = nbr + ((size_t)b * M + q) * KNBR;

    int count = 0;
    for (int i0 = 0; i0 < N && count < KNBR; i0 += 32) {
        int i = i0 + lane;
        float dx = ps[i * 3 + 0] - qx;
        float dy = ps[i * 3 + 1] - qy;
        float dz = ps[i * 3 + 2] - qz;
        float d2 = dx * dx + dy * dy + dz * dz;
        bool inr = d2 <= r2;
        unsigned msk = (unsigned)__ballot(inr);
        int slot = count + __popc(msk & ((1u << lane) - 1u));
        if (inr && slot < KNBR) nb[slot] = i;
        count += __popc(msk);
    }
    if (lane == 0) cnt[(size_t)b * M + q] = count < KNBR ? count : KNBR;
}

// ---------------------------------------------------------------------------
// SetAbstraction MLP + max-aggregation, all matrix math on v_wmma bf16.
//   CF   = feature width of source x
//   CINP = (CF+3) padded to multiple of 32  (WMMA K)
//   CH   = hidden == output channels (multiple of 16)
//   QPB  = queries per block; QPB*(CH/16) must equal 8 waves.
// ---------------------------------------------------------------------------
template <int CF, int CINP, int CH, int QPB>
__global__ void __launch_bounds__(256)
sa_mlp_kernel(const float* __restrict__ xsrc,    // [B, Nsrc, CF]
              const float* __restrict__ possrc,  // [B, Nsrc, 3]
              const float* __restrict__ posq,    // [B, M, 3]
              const int* __restrict__ nbr,       // [B, M, 64]
              const int* __restrict__ cnt,       // [B, M]
              const __bf16* __restrict__ w1p, const float* __restrict__ b1,
              const __bf16* __restrict__ w2p, const float* __restrict__ b2,
              float* __restrict__ out,           // [B, M, CH]
              int M, int Nsrc) {
    constexpr int CT   = CH / 16;
    constexpr int ROWS = QPB * 64;
    constexpr int KT1  = CINP / 32;
    constexpr int KT2  = CH / 32;

    __shared__ __bf16 s_msg[ROWS * CINP];
    __shared__ __bf16 s_h[ROWS * CH];
    __shared__ int    s_nbr[ROWS];
    __shared__ int    s_cnt[QPB];
    __shared__ float  s_pq[QPB][3];

    const int t   = threadIdx.x;
    const int bid = blockIdx.x;
    const int qchunks = M / QPB;
    const int b  = bid / qchunks;
    const int q0 = (bid - b * qchunks) * QPB;

    __builtin_prefetch(w1p, 0, 1);     // global_prefetch_b8: pull weights toward L2/WGP$
    __builtin_prefetch(w2p, 0, 1);

    if (t < ROWS) s_nbr[t] = nbr[((size_t)b * M + q0) * 64 + t];
    if (t < QPB)  s_cnt[t] = cnt[(size_t)b * M + q0 + t];
    if (t >= 32 && t < 32 + QPB * 3) {
        int tt = t - 32;
        s_pq[tt / 3][tt % 3] = posq[((size_t)b * M + q0) * 3 + tt];
    }
    __syncthreads();

    // Build msg = [x_j, pos_j - pos_q] as bf16, zero padded to CINP.
    const float* xb = xsrc + (size_t)b * Nsrc * CF;
    const float* pb = possrc + (size_t)b * Nsrc * 3;
    for (int idx = t; idx < ROWS * CINP; idx += 256) {
        int row = idx / CINP;
        int c   = idx - row * CINP;
        int q   = row >> 6, j = row & 63;
        float v = 0.0f;
        if (j < s_cnt[q]) {
            int si = s_nbr[row];
            if (c < CF)          v = xb[(size_t)si * CF + c];
            else if (c < CF + 3) v = pb[(size_t)si * 3 + (c - CF)] - s_pq[q][c - CF];
        }
        s_msg[idx] = (__bf16)v;
    }
    __syncthreads();

    const int lane  = t & 31, wv = t >> 5;
    const int qh    = wv / CT;            // local query this wave works on
    const int nt    = wv - qh * CT;       // output column tile
    const int col   = lane & 15;
    const int khalf = (lane & 16) ? 8 : 0;

    // ---- GEMM1: H = relu(msg @ W1 + b1) -> LDS (bf16)
    const float bias1 = b1[nt * 16 + col];
#pragma unroll
    for (int rt = 0; rt < 4; ++rt) {
        v8f acc;
#pragma unroll
        for (int r = 0; r < 8; ++r) acc[r] = bias1;
        const int arow = qh * 64 + rt * 16 + col;
#pragma unroll
        for (int kt = 0; kt < KT1; ++kt) {
            const __bf16* ap = &s_msg[arow * CINP + kt * 32 + khalf];
            bf16x8 lo = *(const bf16x8*)ap;
            bf16x8 hi = *(const bf16x8*)(ap + 16);
            bf16x16 a;
#pragma unroll
            for (int e = 0; e < 8; ++e) { a[e] = lo[e]; a[e + 8] = hi[e]; }
            bf16x16 bm = *(const bf16x16*)&w1p[((kt * CT + nt) * 32 + lane) * 16];
            acc = __builtin_amdgcn_wmma_f32_16x16x32_bf16(
                false, a, false, bm, (short)0, acc, false, false);
        }
        const int rbase = qh * 64 + rt * 16 + ((lane & 16) ? 8 : 0);
#pragma unroll
        for (int r = 0; r < 8; ++r) {
            float v = fmaxf(acc[r], 0.0f);
            s_h[(rbase + r) * CH + nt * 16 + col] = (__bf16)v;
        }
    }
    __syncthreads();

    // ---- GEMM2: O = H @ W2 + b2, then masked max over 64 neighbor rows.
    const float bias2 = b2[nt * 16 + col];
    const int   cq    = s_cnt[qh];
    float cmax = -__builtin_inff();
#pragma unroll
    for (int rt = 0; rt < 4; ++rt) {
        v8f acc;
#pragma unroll
        for (int r = 0; r < 8; ++r) acc[r] = bias2;
        const int arow = qh * 64 + rt * 16 + col;
#pragma unroll
        for (int kt = 0; kt < KT2; ++kt) {
            const __bf16* ap = &s_h[arow * CH + kt * 32 + khalf];
            bf16x8 lo = *(const bf16x8*)ap;
            bf16x8 hi = *(const bf16x8*)(ap + 16);
            bf16x16 a;
#pragma unroll
            for (int e = 0; e < 8; ++e) { a[e] = lo[e]; a[e + 8] = hi[e]; }
            bf16x16 bm = *(const bf16x16*)&w2p[((kt * CT + nt) * 32 + lane) * 16];
            acc = __builtin_amdgcn_wmma_f32_16x16x32_bf16(
                false, a, false, bm, (short)0, acc, false, false);
        }
        const int jbase = rt * 16 + ((lane & 16) ? 8 : 0);
#pragma unroll
        for (int r = 0; r < 8; ++r) {
            float v = (jbase + r < cq) ? acc[r] : -__builtin_inff();
            cmax = fmaxf(cmax, v);
        }
    }
    // column n lives in lane n and lane n+16: combine halves.
    cmax = fmaxf(cmax, __shfl_xor(cmax, 16, 32));
    if (cq == 0) cmax = 0.0f;
    if (lane < 16)
        out[((size_t)b * M + q0 + qh) * CH + nt * 16 + col] = cmax;
}

// ---------------------------------------------------------------------------
__global__ void batch_out_kernel(int* __restrict__ bo) {
    int i = blockIdx.x * 256 + threadIdx.x;
    if (i < BATCH * M3s) bo[i] = i / M3s;
}

// ---------------------------------------------------------------------------
extern "C" void kernel_launch(void* const* d_in, const int* in_sizes, int n_in,
                              void* d_out, int out_size, void* d_ws, size_t ws_size,
                              hipStream_t stream) {
    const float* pos = (const float*)d_in[0];
    // d_in[1] = batch (regenerated on output)
    const float* w11 = (const float*)d_in[2];  const float* b11 = (const float*)d_in[3];
    const float* w12 = (const float*)d_in[4];  const float* b12 = (const float*)d_in[5];
    const float* w21 = (const float*)d_in[6];  const float* b21 = (const float*)d_in[7];
    const float* w22 = (const float*)d_in[8];  const float* b22 = (const float*)d_in[9];
    const float* w31 = (const float*)d_in[10]; const float* b31 = (const float*)d_in[11];
    const float* w32 = (const float*)d_in[12]; const float* b32 = (const float*)d_in[13];

    // --- carve workspace ---------------------------------------------------
    char* wp = (char*)d_ws;
    auto take = [&](size_t bytes) {
        char* p = (char*)(((uintptr_t)wp + 255) & ~(uintptr_t)255);
        wp = p + bytes;
        return p;
    };
    __bf16* pk11 = (__bf16*)take(32 * 32 * 2);
    __bf16* pk12 = (__bf16*)take(32 * 32 * 2);
    __bf16* pk21 = (__bf16*)take(64 * 64 * 2);
    __bf16* pk22 = (__bf16*)take(64 * 64 * 2);
    __bf16* pk31 = (__bf16*)take(96 * 128 * 2);
    __bf16* pk32 = (__bf16*)take(128 * 128 * 2);
    float* pos1 = (float*)take((size_t)BATCH * M1s * 3 * 4);
    float* pos2 = (float*)take((size_t)BATCH * M2s * 3 * 4);
    int*   nbr1 = (int*)take((size_t)BATCH * M1s * KNBR * 4);
    int*   cnt1 = (int*)take((size_t)BATCH * M1s * 4);
    int*   nbr2 = (int*)take((size_t)BATCH * M2s * KNBR * 4);
    int*   cnt2 = (int*)take((size_t)BATCH * M2s * 4);
    int*   nbr3 = (int*)take((size_t)BATCH * M3s * KNBR * 4);
    int*   cnt3 = (int*)take((size_t)BATCH * M3s * 4);
    float* x1   = (float*)take((size_t)BATCH * M1s * 32 * 4);
    float* x2   = (float*)take((size_t)BATCH * M2s * 64 * 4);

    // --- outputs: x3 [8,512,128] | p3 [8,512,3] | batch [8,512] ------------
    float* x3o = (float*)d_out;
    float* p3o = x3o + (size_t)BATCH * M3s * 128;
    int*   bo  = (int*)(p3o + (size_t)BATCH * M3s * 3);

    // --- pack weights into WMMA B-fragment layout --------------------------
    pack_w_kernel<<<8, 256, 0, stream>>>(w11, pk11, 6,   32,  32);
    pack_w_kernel<<<8, 256, 0, stream>>>(w12, pk12, 32,  32,  32);
    pack_w_kernel<<<8, 256, 0, stream>>>(w21, pk21, 35,  64,  64);
    pack_w_kernel<<<8, 256, 0, stream>>>(w22, pk22, 64,  64,  64);
    pack_w_kernel<<<16, 256, 0, stream>>>(w31, pk31, 67, 96,  128);
    pack_w_kernel<<<16, 256, 0, stream>>>(w32, pk32, 128, 128, 128);

    // --- layer 1: 4096 -> 2048, r=0.2, (3+3)->32->32 -----------------------
    fps_kernel<4096, 2048><<<BATCH, 256, 0, stream>>>(pos, pos1);
    ballq_kernel<<<BATCH * M1s / 8, 256, 0, stream>>>(pos, pos1, nbr1, cnt1,
                                                      4096, M1s, 0.04f);
    sa_mlp_kernel<3, 32, 32, 4><<<BATCH * M1s / 4, 256, 0, stream>>>(
        pos, pos, pos1, nbr1, cnt1, pk11, b11, pk12, b12, x1, M1s, 4096);

    // --- layer 2: 2048 -> 512, r=0.4, (32+3)->64->64 -----------------------
    fps_kernel<2048, 512><<<BATCH, 256, 0, stream>>>(pos1, pos2);
    ballq_kernel<<<BATCH * M2s / 8, 256, 0, stream>>>(pos1, pos2, nbr2, cnt2,
                                                      2048, M2s, 0.16f);
    sa_mlp_kernel<32, 64, 64, 2><<<BATCH * M2s / 2, 256, 0, stream>>>(
        x1, pos1, pos2, nbr2, cnt2, pk21, b21, pk22, b22, x2, M2s, 2048);

    // --- layer 3: 512 -> 512, r=1.0, (64+3)->128->128 ----------------------
    fps_kernel<512, 512><<<BATCH, 256, 0, stream>>>(pos2, p3o);
    ballq_kernel<<<BATCH * M3s / 8, 256, 0, stream>>>(pos2, p3o, nbr3, cnt3,
                                                      512, M3s, 1.0f);
    sa_mlp_kernel<64, 96, 128, 1><<<BATCH * M3s, 256, 0, stream>>>(
        x2, pos2, p3o, nbr3, cnt3, pk31, b31, pk32, b32, x3o, M3s, 512);

    batch_out_kernel<<<(BATCH * M3s + 255) / 256, 256, 0, stream>>>(bo);
}
